// MultiHeadAttentionWReduction_81028853006755
// MI455X (gfx1250) — compile-verified, matrix operand order checked by reading, not run on hardware
//
#include <hip/hip_runtime.h>

// Problem constants (B=2, S=2048, D=1024, H=16, HD=64)
#define BB 2
#define SS 2048
#define DD 1024
#define HH 16
#define HD 64

typedef __bf16 bf16_t;
typedef __attribute__((ext_vector_type(16))) __bf16 bf16x16;
typedef __attribute__((ext_vector_type(8)))  __bf16 bf16x8;
typedef __attribute__((ext_vector_type(8)))  float  f32x8;

union ABFrag {
    bf16x16 v;
    bf16x8  h[2];
};

static __device__ __forceinline__ f32x8 wmma_bf16(bf16x16 a, bf16x16 b, f32x8 c) {
    // D = A(16x32 bf16) * B(32x16 bf16) + C(16x16 f32)
    return __builtin_amdgcn_wmma_f32_16x16x32_bf16(false, a, false, b,
                                                   (short)0, c, false, false);
}

// Async global -> LDS, 16 bytes per lane, tracked with ASYNCcnt (CDNA5 path).
// LDS destination = low 32 bits of the shared-aperture flat address
// (ISA 10.2).
static __device__ __forceinline__ void async_b128(const bf16_t* lds_ptr,
                                                  const bf16_t* gptr) {
    uint32_t lds_off = (uint32_t)(uintptr_t)lds_ptr;
    asm volatile("global_load_async_to_lds_b128 %0, %1, off"
                 :: "v"(lds_off), "v"(gptr) : "memory");
}

static __device__ __forceinline__ void wait_async0() {
    asm volatile("s_wait_asynccnt 0" ::: "memory");
}

// ---------------------------------------------------------------------------
// fp32 -> bf16 conversion, 8 elements / thread, fully vectorized
// ---------------------------------------------------------------------------
__global__ __launch_bounds__(256) void cvt_f32_bf16(const float* __restrict__ in,
                                                    bf16_t* __restrict__ out) {
    int i = (blockIdx.x * 256 + threadIdx.x) * 8;
    float4 a = *(const float4*)(in + i);
    float4 c = *(const float4*)(in + i + 4);
    bf16x8 o;
    o[0] = (bf16_t)a.x; o[1] = (bf16_t)a.y; o[2] = (bf16_t)a.z; o[3] = (bf16_t)a.w;
    o[4] = (bf16_t)c.x; o[5] = (bf16_t)c.y; o[6] = (bf16_t)c.z; o[7] = (bf16_t)c.w;
    *(bf16x8*)(out + i) = o;
}

// ---------------------------------------------------------------------------
// GEMM: out[m,n] = sum_k A[m,k] * W[n,k] + bias[n]   (A @ W^T)
// Block 256 thr (8 waves), tile 128x128, K panels of 32 double-buffered in
// LDS via async loads. Wave w: rows (w&3)*32 (two 16-row m-tiles), cols
// (w>>2)*64 (four 16-col n-tiles) -> 8 WMMA / K-step on 6 fragment loads.
// mode 0: fp32 out [M,N]       (final projection -> d_out)
// mode 1: bf16 out [B,H,S,HD]  (Q, K head layout)
// mode 2: bf16 out [B,H,HD,S]  (V transposed head layout)
// ---------------------------------------------------------------------------
__global__ __launch_bounds__(256) void gemm_wmma(const bf16_t* __restrict__ A,
                                                 const bf16_t* __restrict__ W,
                                                 const float* __restrict__ bias,
                                                 void* __restrict__ out,
                                                 int mode) {
    __shared__ bf16_t sA[2][128 * 40];   // stride 40 elems = 80 B (5 x 16 B)
    __shared__ bf16_t sB[2][128 * 40];

    const int tid  = threadIdx.x;
    const int wave = tid >> 5;
    const int lane = tid & 31;
    const int ln   = lane & 15;
    const int hf   = lane >> 4;
    const int kh   = hf * 8;
    const int wm   = wave & 3;
    const int wn   = wave >> 2;
    const int rowBase = blockIdx.x * 128;
    const int colBase = blockIdx.y * 128;

    const int srow = tid >> 2;            // 0..63
    const int sch  = (tid & 3) * 8;       // chunk of 8 bf16 (16 B)

    // stage one 128x32 A panel + 128x32 W panel into buffer `buf`
    auto stage = [&](int buf, int k0) {
        async_b128(&sA[buf][srow * 40 + sch],        A + (rowBase + srow) * DD + k0 + sch);
        async_b128(&sA[buf][(srow + 64) * 40 + sch], A + (rowBase + srow + 64) * DD + k0 + sch);
        async_b128(&sB[buf][srow * 40 + sch],        W + (colBase + srow) * DD + k0 + sch);
        async_b128(&sB[buf][(srow + 64) * 40 + sch], W + (colBase + srow + 64) * DD + k0 + sch);
    };

    f32x8 acc[2][4] = {};

    stage(0, 0);
    wait_async0();
    __syncthreads();

    int buf = 0;
    for (int k0 = 0; k0 < DD; k0 += 32) {
        if (k0 + 32 < DD) stage(buf ^ 1, k0 + 32);

        ABFrag a[2], b[4];
#pragma unroll
        for (int mi = 0; mi < 2; ++mi) {
            const bf16_t* p = &sA[buf][(wm * 32 + mi * 16 + ln) * 40];
            a[mi].h[0] = *(const bf16x8*)(p + kh);
            a[mi].h[1] = *(const bf16x8*)(p + 16 + kh);
        }
#pragma unroll
        for (int j = 0; j < 4; ++j) {
            const bf16_t* p = &sB[buf][(wn * 64 + j * 16 + ln) * 40];
            b[j].h[0] = *(const bf16x8*)(p + kh);
            b[j].h[1] = *(const bf16x8*)(p + 16 + kh);
        }
#pragma unroll
        for (int mi = 0; mi < 2; ++mi)
#pragma unroll
            for (int j = 0; j < 4; ++j)
                acc[mi][j] = wmma_bf16(a[mi].v, b[j].v, acc[mi][j]);

        wait_async0();       // my async stores into buf^1 are in LDS
        __syncthreads();     // (compiler also waits DScnt for my reads of buf)
        buf ^= 1;
    }

    // Epilogue. C layout: lanes 0-15 hold M=i, lanes 16-31 hold M=i+8, N=ln.
#pragma unroll
    for (int mi = 0; mi < 2; ++mi) {
#pragma unroll
        for (int j = 0; j < 4; ++j) {
            int col = colBase + wn * 64 + j * 16 + ln;
            float bv = bias[col];
#pragma unroll
            for (int i = 0; i < 8; ++i) {
                int row = rowBase + wm * 32 + mi * 16 + i + 8 * hf;
                float val = acc[mi][j][i] + bv;
                if (mode == 0) {
                    ((float*)out)[row * DD + col] = val;
                } else {
                    int hh = col >> 6, dd = col & 63;
                    int bb = row >> 11, ss = row & 2047;
                    if (mode == 1)
                        ((bf16_t*)out)[(((bb * HH + hh) * SS + ss) << 6) + dd] = (bf16_t)val;
                    else
                        ((bf16_t*)out)[(((bb * HH + hh) << 6) + dd) * SS + ss] = (bf16_t)val;
                }
            }
        }
    }
}

// ---------------------------------------------------------------------------
// Flash attention. Grid (S/128, B*H), 256 thr = 8 waves; each wave owns a
// 16-query tile. Per 32-key tile: K (32x64) and V^T (64x32) are staged ONCE
// per block into double-buffered LDS via async loads (8x less L2 traffic than
// per-wave loads), then each wave: S = Q K^T (4 wmma), online softmax
// (shuffle row-reductions in 16-lane halves matching the C layout), P goes
// C-layout -> per-wave LDS -> A-layout, O += P V (4 wmma).
// ---------------------------------------------------------------------------
__global__ __launch_bounds__(256) void flash_attn(const bf16_t* __restrict__ Q,
                                                  const bf16_t* __restrict__ Kc,
                                                  const bf16_t* __restrict__ VT,
                                                  bf16_t* __restrict__ AO) {
    __shared__ bf16_t sK[2][32 * 72];   // key-major 32x64, stride 72 (144 B)
    __shared__ bf16_t sV[2][64 * 40];   // d-major 64x32, stride 40 (80 B)
    __shared__ bf16_t sp[8][16 * 32];   // per-wave P tile

    const int tid  = threadIdx.x;
    const int wave = tid >> 5;
    const int lane = tid & 31;
    const int ln   = lane & 15;
    const int hf   = lane >> 4;
    const int kh   = hf * 8;
    const int bh   = blockIdx.y;
    const int bb   = bh >> 4;
    const int hh   = bh & 15;

    const bf16_t* q  = Q  + (size_t)bh * SS * HD;
    const bf16_t* k  = Kc + (size_t)bh * SS * HD;
    const bf16_t* vt = VT + (size_t)bh * HD * SS;
    const int qrow = blockIdx.x * 128 + wave * 16 + ln;

    const int krow = tid >> 3, kch = (tid & 7) * 8;   // 32 rows x 8 chunks
    const int vrow = tid >> 2, vch = (tid & 3) * 8;   // 64 rows x 4 chunks
    auto stage = [&](int buf, int kt) {
        async_b128(&sK[buf][krow * 72 + kch], k + (kt * 32 + krow) * HD + kch);
        async_b128(&sV[buf][vrow * 40 + vch], vt + vrow * SS + kt * 32 + vch);
    };

    // Q fragments for the full head dim (K = 64 -> two 16x32 A fragments)
    ABFrag aq[2];
#pragma unroll
    for (int t = 0; t < 2; ++t) {
        aq[t].h[0] = *(const bf16x8*)(q + qrow * HD + t * 32 + kh);
        aq[t].h[1] = *(const bf16x8*)(q + qrow * HD + t * 32 + 16 + kh);
    }

    f32x8 o[4] = {};
    float mst[8], lst[8];
#pragma unroll
    for (int i = 0; i < 8; ++i) { mst[i] = -1e30f; lst[i] = 0.0f; }

    bf16_t* myp = &sp[wave][0];
    const float scale = 0.125f;   // 1/sqrt(64)

    stage(0, 0);
    wait_async0();
    __syncthreads();

    int buf = 0;
    for (int kt = 0; kt < SS / 32; ++kt) {
        if (kt + 1 < SS / 32) stage(buf ^ 1, kt + 1);

        // ---- S tile = Q (16x64) @ K^T (64x32) ----
        f32x8 s0 = {}, s1 = {};
#pragma unroll
        for (int t = 0; t < 2; ++t) {
            ABFrag bk0, bk1;
            const bf16_t* p0 = &sK[buf][ln * 72];
            const bf16_t* p1 = &sK[buf][(16 + ln) * 72];
            bk0.h[0] = *(const bf16x8*)(p0 + t * 32 + kh);
            bk0.h[1] = *(const bf16x8*)(p0 + t * 32 + 16 + kh);
            bk1.h[0] = *(const bf16x8*)(p1 + t * 32 + kh);
            bk1.h[1] = *(const bf16x8*)(p1 + t * 32 + 16 + kh);
            s0 = wmma_bf16(aq[t].v, bk0.v, s0);
            s1 = wmma_bf16(aq[t].v, bk1.v, s1);
        }

        // ---- online softmax: row reductions across 16-lane halves ----
#pragma unroll
        for (int i = 0; i < 8; ++i) {
            float v0 = s0[i] * scale, v1 = s1[i] * scale;
            float mx = fmaxf(v0, v1);
            mx = fmaxf(mx, __shfl_xor(mx, 1, 32));
            mx = fmaxf(mx, __shfl_xor(mx, 2, 32));
            mx = fmaxf(mx, __shfl_xor(mx, 4, 32));
            mx = fmaxf(mx, __shfl_xor(mx, 8, 32));
            float nm   = fmaxf(mst[i], mx);
            float corr = __expf(mst[i] - nm);
            float p0 = __expf(v0 - nm), p1 = __expf(v1 - nm);
            float rs = p0 + p1;
            rs += __shfl_xor(rs, 1, 32);
            rs += __shfl_xor(rs, 2, 32);
            rs += __shfl_xor(rs, 4, 32);
            rs += __shfl_xor(rs, 8, 32);
            lst[i] = lst[i] * corr + rs;
            mst[i] = nm;
#pragma unroll
            for (int j = 0; j < 4; ++j) o[j][i] *= corr;
            int prow = i + 8 * hf;
            myp[prow * 32 + ln]      = (bf16_t)p0;
            myp[prow * 32 + 16 + ln] = (bf16_t)p1;
        }
        asm volatile("" ::: "memory");   // same-wave DS ops are HW in-order

        // ---- O += P (16x32) @ V (32x64) ----
        ABFrag pf;
        pf.h[0] = *(const bf16x8*)(myp + ln * 32 + kh);
        pf.h[1] = *(const bf16x8*)(myp + ln * 32 + 16 + kh);
#pragma unroll
        for (int j = 0; j < 4; ++j) {
            ABFrag vf;
            const bf16_t* p = &sV[buf][(j * 16 + ln) * 40];
            vf.h[0] = *(const bf16x8*)(p + kh);
            vf.h[1] = *(const bf16x8*)(p + 16 + kh);
            o[j] = wmma_bf16(pf.v, vf.v, o[j]);
        }

        wait_async0();
        __syncthreads();
        buf ^= 1;
    }

    // ---- normalize and write bf16 [B,S,D] for the final projection GEMM ----
#pragma unroll
    for (int j = 0; j < 4; ++j) {
        int d = j * 16 + ln;
#pragma unroll
        for (int i = 0; i < 8; ++i) {
            int sr = blockIdx.x * 128 + wave * 16 + i + 8 * hf;
            float val = o[j][i] / lst[i];
            AO[(size_t)(bb * SS + sr) * DD + hh * HD + d] = (bf16_t)val;
        }
    }
}

// ---------------------------------------------------------------------------
extern "C" void kernel_launch(void* const* d_in, const int* in_sizes, int n_in,
                              void* d_out, int out_size, void* d_ws, size_t ws_size,
                              hipStream_t stream) {
    const float* x  = (const float*)d_in[0];
    const float* Wk = (const float*)d_in[1];
    const float* bk = (const float*)d_in[2];
    const float* Wq = (const float*)d_in[3];
    const float* bq = (const float*)d_in[4];
    const float* Wv = (const float*)d_in[5];
    const float* bv = (const float*)d_in[6];
    const float* Wo = (const float*)d_in[7];
    const float* bo = (const float*)d_in[8];

    char* ws = (char*)d_ws;
    const size_t MB = 1024u * 1024u;
    bf16_t* xb  = (bf16_t*)(ws + 0);        // [4096,1024]       8 MB
    bf16_t* wqb = (bf16_t*)(ws + 8 * MB);   // [1024,1024]       2 MB
    bf16_t* wkb = (bf16_t*)(ws + 10 * MB);
    bf16_t* wvb = (bf16_t*)(ws + 12 * MB);
    bf16_t* wob = (bf16_t*)(ws + 14 * MB);
    bf16_t* qb  = (bf16_t*)(ws + 16 * MB);  // [B,H,S,HD]        8 MB
    bf16_t* kb  = (bf16_t*)(ws + 24 * MB);  // [B,H,S,HD]        8 MB
    bf16_t* vtb = (bf16_t*)(ws + 32 * MB);  // [B,H,HD,S]        8 MB
    bf16_t* ab  = (bf16_t*)(ws + 40 * MB);  // [B,S,D] bf16      8 MB

    cvt_f32_bf16<<<2048, 256, 0, stream>>>(x, xb);
    cvt_f32_bf16<<<512, 256, 0, stream>>>(Wq, wqb);
    cvt_f32_bf16<<<512, 256, 0, stream>>>(Wk, wkb);
    cvt_f32_bf16<<<512, 256, 0, stream>>>(Wv, wvb);
    cvt_f32_bf16<<<512, 256, 0, stream>>>(Wo, wob);

    dim3 gg(4096 / 128, 1024 / 128);        // 32 x 8 blocks
    gemm_wmma<<<gg, 256, 0, stream>>>(xb, wqb, bq, qb, 1);
    gemm_wmma<<<gg, 256, 0, stream>>>(xb, wkb, bk, kb, 1);
    gemm_wmma<<<gg, 256, 0, stream>>>(xb, wvb, bv, vtb, 2);

    flash_attn<<<dim3(SS / 128, BB * HH), 256, 0, stream>>>(qb, kb, vtb, ab);

    gemm_wmma<<<gg, 256, 0, stream>>>(ab, wob, bo, d_out, 0);
}
